// LSTM_4655744548934
// MI455X (gfx1250) — matrix operand order repaired and checked by reading, not compiled, not run on hardware
//
#include <hip/hip_runtime.h>
#include <hip/hip_bf16.h>

// Problem constants (from reference)
#define B_      256
#define SEQ_    256
#define INSZ    128
#define HID     1024
#define NG      4096      // 4*HID
#define ZS      1026
#define INDIM   1154      // 2 + 128 + 1024
#define KPAD    1184      // 37 * 32 (pad K for 16x16x32 WMMA)
#define KTILES  37

typedef __attribute__((ext_vector_type(16))) __bf16 v16bf;
typedef __attribute__((ext_vector_type(8)))  __bf16 v8bf;
typedef __attribute__((ext_vector_type(8)))  float  v8f;

__device__ __forceinline__ float sigf(float x) { return 1.0f / (1.0f + __expf(-x)); }

// ---------------------------------------------------------------------------
// One-time: convert WU_w (4096 x 1154 f32, row-major) to bf16, K-padded to 1184
// ---------------------------------------------------------------------------
__global__ __launch_bounds__(256) void k_init_wu(const float* __restrict__ W,
                                                 __bf16* __restrict__ WUb) {
  int idx = blockIdx.x * 256 + threadIdx.x;           // 0 .. NG*KPAD-1
  int j = idx / KPAD;
  int k = idx - j * KPAD;
  float v = (k < INDIM) ? W[(size_t)j * INDIM + k] : 0.0f;
  WUb[idx] = (__bf16)v;
}

// ---------------------------------------------------------------------------
// One-time: build bf16 input vector for t=0 from z0/u0, init c and x buffers
// ---------------------------------------------------------------------------
__global__ __launch_bounds__(256) void k_init_state(const float* __restrict__ z0,
                                                    const float* __restrict__ cz0,
                                                    const float* __restrict__ u,
                                                    __bf16* __restrict__ inpB,
                                                    float* __restrict__ cbuf,
                                                    float* __restrict__ xbuf) {
  int idx = blockIdx.x * 256 + threadIdx.x;           // 0 .. B_*KPAD-1
  int b = idx / KPAD;
  int k = idx - b * KPAD;
  float v;
  if (k < 2) {
    v = z0[(size_t)b * ZS + k];
    xbuf[b * 2 + k] = v;
  } else if (k < 2 + INSZ) {
    v = u[((size_t)b * SEQ_ + 0) * INSZ + (k - 2)];
  } else if (k < INDIM) {
    v = z0[(size_t)b * ZS + 2 + (k - 2 - INSZ)];
  } else {
    v = 0.0f;
  }
  inpB[idx] = (__bf16)v;
  if (k < HID) cbuf[(size_t)b * HID + k] = cz0[(size_t)b * ZS + 2 + k];
}

// ---------------------------------------------------------------------------
// Per-step GEMM: gates(256 x 4096, f32) = inpB(256 x KPAD, bf16) @ WUb^T
// Block = 8 waves; block tile 32(M) x 256(N). Each wave owns a 32x32 footprint
// (2 M-subtiles x 2 N-strips) -> 4 WMMA accumulators; A and B fragments are
// each reused twice per K-step (2 b128 loads per WMMA).
// ---------------------------------------------------------------------------
__global__ __launch_bounds__(256) void k_gemm_gates(const __bf16* __restrict__ inpB,
                                                    const __bf16* __restrict__ WUb,
                                                    float* __restrict__ gates) {
  const int lane = threadIdx.x & 31;
  const int wave = threadIdx.x >> 5;
  const int mBase = (blockIdx.x & 7) * 32;    // 8 M-blocks * 32 rows  = 256
  const int nBase = (blockIdx.x >> 3) * 256;  // 16 N-blocks * 256 cols = 4096
  const int hf  = lane >> 4;                  // lane half (0/1)
  const int l16 = lane & 15;

  // B fragments: two 16-col strips; per-lane 16 contiguous K bf16 (32B aligned)
  const int n0 = nBase + wave * 32 + l16;     // strip 0
  const __bf16* pb0 = WUb + (size_t)n0 * KPAD + hf * 16;
  const __bf16* pb1 = pb0 + (size_t)16 * KPAD;  // strip 1 (n0 + 16)

  // A fragments: two M sub-tiles (rows of inp); kbase = 16*hf (contiguous, matches B)
  const int m0 = mBase + l16;
  const __bf16* pa0 = inpB + (size_t)m0 * KPAD + hf * 16;
  const __bf16* pa1 = pa0 + (size_t)16 * KPAD;

  v8f acc00 = {};   // (m0 tile, n strip 0)
  v8f acc01 = {};   // (m0 tile, n strip 1)
  v8f acc10 = {};   // (m1 tile, n strip 0)
  v8f acc11 = {};   // (m1 tile, n strip 1)

  for (int kt = 0; kt < KTILES; ++kt) {
    const int k0 = kt * 32;
    __builtin_prefetch((const void*)(pb0 + k0 + 64), 0, 1);   // global_prefetch_b8
    __builtin_prefetch((const void*)(pb1 + k0 + 64), 0, 1);

    v16bf bf0 = *(const v16bf*)(pb0 + k0);
    v16bf bf1 = *(const v16bf*)(pb1 + k0);

    v8bf a0lo = *(const v8bf*)(pa0 + k0);
    v8bf a0hi = *(const v8bf*)(pa0 + k0 + 8);
    v8bf a1lo = *(const v8bf*)(pa1 + k0);
    v8bf a1hi = *(const v8bf*)(pa1 + k0 + 8);
    v16bf af0, af1;
#pragma unroll
    for (int i = 0; i < 8; ++i) {
      af0[i] = a0lo[i]; af0[8 + i] = a0hi[i];
      af1[i] = a1lo[i]; af1[8 + i] = a1hi[i];
    }
    acc00 = __builtin_amdgcn_wmma_f32_16x16x32_bf16(false, af0, false, bf0,
                                                    (short)0, acc00, false, false);
    acc10 = __builtin_amdgcn_wmma_f32_16x16x32_bf16(false, af1, false, bf0,
                                                    (short)0, acc10, false, false);
    acc01 = __builtin_amdgcn_wmma_f32_16x16x32_bf16(false, af0, false, bf1,
                                                    (short)0, acc01, false, false);
    acc11 = __builtin_amdgcn_wmma_f32_16x16x32_bf16(false, af1, false, bf1,
                                                    (short)0, acc11, false, false);
  }

  // C/D layout: reg r -> row (base + r + 8*hf), col = strip base + l16
  float* g0 = gates + (size_t)(mBase + 8 * hf) * NG + n0;
#pragma unroll
  for (int r = 0; r < 8; ++r) {
    g0[(size_t)r * NG]             = acc00[r];
    g0[(size_t)r * NG + 16]        = acc01[r];
    g0[(size_t)(r + 16) * NG]      = acc10[r];
    g0[(size_t)(r + 16) * NG + 16] = acc11[r];
  }
}

// ---------------------------------------------------------------------------
// Per-step LSTM cell: c,h update; writes h into output slab and next-step inpB
// ---------------------------------------------------------------------------
__global__ __launch_bounds__(256) void k_cell(const float* __restrict__ gates,
                                              const float* __restrict__ WU_b,
                                              float* __restrict__ cbuf,
                                              float* __restrict__ hbuf,
                                              __bf16* __restrict__ inpB,
                                              float* __restrict__ out, int t) {
  int idx = blockIdx.x * 256 + threadIdx.x;  // 0 .. B_*HID-1
  int b = idx >> 10;
  int j = idx & (HID - 1);
  const float* g = gates + (size_t)b * NG + j;
  float ig = g[0]        + WU_b[j];
  float fg = g[HID]      + WU_b[HID + j];
  float gg = g[2 * HID]  + WU_b[2 * HID + j];
  float og = g[3 * HID]  + WU_b[3 * HID + j];
  float cn = sigf(fg) * cbuf[idx] + sigf(ig) * tanhf(gg);
  float hn = sigf(og) * tanhf(cn);
  cbuf[idx] = cn;
  hbuf[idx] = hn;
  out[((size_t)b * SEQ_ + t) * ZS + 2 + j] = hn;
  inpB[(size_t)b * KPAD + 2 + INSZ + j] = (__bf16)hn;  // h for next step
}

// ---------------------------------------------------------------------------
// Per-step small state update: one wave per batch element.
// alpha = sig(h@alpha_w^T + b), hx = h@Whx_w^T + b, x_mid = (I + tau*A) x_prev,
// x_next = alpha*x_mid + (1-alpha)*hx. Also stages u_{t+1} into inpB.
// ---------------------------------------------------------------------------
__global__ __launch_bounds__(256) void k_xstate(const float* __restrict__ hbuf,
                                                const float* __restrict__ alpha_w,
                                                const float* __restrict__ alpha_b,
                                                const float* __restrict__ Whx_w,
                                                const float* __restrict__ Whx_b,
                                                const float* __restrict__ tau,
                                                const float* __restrict__ rnn_input,
                                                float* __restrict__ xbuf,
                                                __bf16* __restrict__ inpB,
                                                float* __restrict__ out, int t) {
  const int lane = threadIdx.x & 31;
  const int wave = threadIdx.x >> 5;
  const int b = blockIdx.x * 8 + wave;
  const float* h = hbuf + (size_t)b * HID;

  float d0 = 0.f, d1 = 0.f, d2 = 0.f, d3 = 0.f;
  for (int j = lane; j < HID; j += 32) {
    float hv = h[j];
    d0 += hv * alpha_w[j];
    d1 += hv * alpha_w[HID + j];
    d2 += hv * Whx_w[j];
    d3 += hv * Whx_w[HID + j];
  }
#pragma unroll
  for (int s = 16; s > 0; s >>= 1) {
    d0 += __shfl_xor(d0, s, 32);
    d1 += __shfl_xor(d1, s, 32);
    d2 += __shfl_xor(d2, s, 32);
    d3 += __shfl_xor(d3, s, 32);
  }

  if (lane == 0) {
    const float MHU = 1.5f;
    float a0  = sigf(d0 + alpha_b[0]);
    float a1  = sigf(d1 + alpha_b[1]);
    float hx0 = d2 + Whx_b[0];
    float hx1 = d3 + Whx_b[1];
    float ta  = tau[(size_t)b * SEQ_ + t];
    float x0  = xbuf[b * 2 + 0];
    float x1  = xbuf[b * 2 + 1];
    // M = I + tau*A, A = [[MHU,-MHU],[1/MHU,0]]
    float xm0 = (1.0f + ta * MHU) * x0 + (-ta * MHU) * x1;
    float xm1 = (ta / MHU) * x0 + x1;
    float xn0 = a0 * xm0 + (1.0f - a0) * hx0;
    float xn1 = a1 * xm1 + (1.0f - a1) * hx1;
    xbuf[b * 2 + 0] = xn0;
    xbuf[b * 2 + 1] = xn1;
    out[((size_t)b * SEQ_ + t) * ZS + 0] = xn0;
    out[((size_t)b * SEQ_ + t) * ZS + 1] = xn1;
    inpB[(size_t)b * KPAD + 0] = (__bf16)xn0;
    inpB[(size_t)b * KPAD + 1] = (__bf16)xn1;
  }

  if (t + 1 < SEQ_) {  // stage u_{t+1} for next step's GEMM input
    const float* u = rnn_input + ((size_t)b * SEQ_ + (t + 1)) * INSZ;
    for (int q = lane; q < INSZ; q += 32)
      inpB[(size_t)b * KPAD + 2 + q] = (__bf16)u[q];
  }
}

// ---------------------------------------------------------------------------
// Final: emit z_f = [x_final, h_final] and cz_f = [c_x(from c_z0), c_final]
// ---------------------------------------------------------------------------
__global__ __launch_bounds__(256) void k_finalize(const float* __restrict__ xbuf,
                                                  const float* __restrict__ hbuf,
                                                  const float* __restrict__ cbuf,
                                                  const float* __restrict__ cz0,
                                                  float* __restrict__ out) {
  int idx = blockIdx.x * 256 + threadIdx.x;  // 0 .. B_*ZS-1
  int b = idx / ZS;
  int q = idx - b * ZS;
  const size_t O1 = (size_t)B_ * SEQ_ * ZS;
  const size_t O2 = O1 + (size_t)B_ * ZS;
  float zv = (q < 2) ? xbuf[b * 2 + q] : hbuf[(size_t)b * HID + (q - 2)];
  float cv = (q < 2) ? cz0[(size_t)b * ZS + q] : cbuf[(size_t)b * HID + (q - 2)];
  out[O1 + (size_t)b * ZS + q] = zv;
  out[O2 + (size_t)b * ZS + q] = cv;
}

// ---------------------------------------------------------------------------
extern "C" void kernel_launch(void* const* d_in, const int* in_sizes, int n_in,
                              void* d_out, int out_size, void* d_ws, size_t ws_size,
                              hipStream_t stream) {
  const float* rnn_input = (const float*)d_in[0];
  const float* tau       = (const float*)d_in[1];
  const float* z0        = (const float*)d_in[2];
  const float* c_z0      = (const float*)d_in[3];
  const float* WU_w      = (const float*)d_in[4];
  const float* WU_b      = (const float*)d_in[5];
  const float* alpha_w   = (const float*)d_in[6];
  const float* alpha_b   = (const float*)d_in[7];
  const float* Whx_w     = (const float*)d_in[8];
  const float* Whx_b     = (const float*)d_in[9];
  float* out = (float*)d_out;

  // Workspace carve-up (all offsets 256B aligned). Total ~16.6 MB.
  char* ws = (char*)d_ws;
  constexpr size_t SZ_WUB   = (size_t)NG * KPAD * sizeof(__bf16);   // 9,699,328
  constexpr size_t SZ_INPB  = (size_t)B_ * KPAD * sizeof(__bf16);  //   606,208
  constexpr size_t SZ_GATES = (size_t)B_ * NG * sizeof(float);     // 4,194,304
  constexpr size_t SZ_H     = (size_t)B_ * HID * sizeof(float);    // 1,048,576
  constexpr size_t SZ_C     = SZ_H;
  __bf16* WUb   = (__bf16*)(ws);
  __bf16* inpB  = (__bf16*)(ws + SZ_WUB);
  float*  gates = (float*)(ws + SZ_WUB + SZ_INPB);
  float*  hbuf  = (float*)(ws + SZ_WUB + SZ_INPB + SZ_GATES);
  float*  cbuf  = (float*)(ws + SZ_WUB + SZ_INPB + SZ_GATES + SZ_H);
  float*  xbuf  = (float*)(ws + SZ_WUB + SZ_INPB + SZ_GATES + SZ_H + SZ_C);

  // One-time setup (weights to bf16, t=0 input vector, c/x state)
  k_init_wu<<<(NG * KPAD) / 256, 256, 0, stream>>>(WU_w, WUb);
  k_init_state<<<(B_ * KPAD) / 256, 256, 0, stream>>>(z0, c_z0, rnn_input,
                                                      inpB, cbuf, xbuf);

  // Sequential scan: GEMM (WMMA) -> cell -> small state update
  for (int t = 0; t < SEQ_; ++t) {
    k_gemm_gates<<<128, 256, 0, stream>>>(inpB, WUb, gates);
    k_cell<<<(B_ * HID) / 256, 256, 0, stream>>>(gates, WU_b, cbuf, hbuf,
                                                 inpB, out, t);
    k_xstate<<<B_ / 8, 256, 0, stream>>>(hbuf, alpha_w, alpha_b, Whx_w, Whx_b,
                                         tau, rnn_input, xbuf, inpB, out, t);
  }

  k_finalize<<<(B_ * ZS + 255) / 256, 256, 0, stream>>>(xbuf, hbuf, cbuf,
                                                        c_z0, out);
}
